// AlexNet_CIFAR10_dsmil_3573412790874
// MI455X (gfx1250) — matrix operand order repaired and checked by reading, not compiled, not run on hardware
//
#include <hip/hip_runtime.h>
#include <hip/hip_bf16.h>
#include <math.h>

// ---------------------------------------------------------------------------
// DSMIL head for MI455X (gfx1250): bf16 WMMA GEMMs with async-LDS double
// buffering + small reduction kernels.
// ---------------------------------------------------------------------------

typedef __attribute__((ext_vector_type(16))) __bf16 v16bf;
typedef __attribute__((ext_vector_type(8)))  float  v8f;
typedef __attribute__((ext_vector_type(8)))  int    v8i;
typedef int v4i_vs __attribute__((vector_size(16)));   // matches builtin param type

#define AS1 __attribute__((address_space(1)))
#define AS3 __attribute__((address_space(3)))

#define N_INST 32768
#define D_IN   1728
#define DMODEL 1024

#if __has_builtin(__builtin_amdgcn_global_load_async_to_lds_b128)
#define USE_ASYNC_LDS 1
#else
#define USE_ASYNC_LDS 0
#endif

static __device__ __forceinline__ unsigned short f2bf(float f) {
  union { float f; unsigned int u; } x; x.f = f;
  unsigned int u = x.u;
  unsigned int r = u + 0x7FFFu + ((u >> 16) & 1u);   // round-to-nearest-even
  return (unsigned short)(r >> 16);
}
static __device__ __forceinline__ float bf2f(unsigned short s) {
  union { unsigned int u; float f; } x; x.u = ((unsigned int)s) << 16; return x.f;
}

// ---------------------------------------------------------------------------
// fp32 -> bf16 conversion, 4 elements per thread (all sizes here are /4)
// ---------------------------------------------------------------------------
__global__ void cvt_f32_bf16_v4(const float* __restrict__ src,
                                unsigned short* __restrict__ dst, int n4) {
  int i = blockIdx.x * blockDim.x + threadIdx.x;
  if (i >= n4) return;
  float4 f = ((const float4*)src)[i];
  uint2 r;
  r.x = (unsigned int)f2bf(f.x) | ((unsigned int)f2bf(f.y) << 16);
  r.y = (unsigned int)f2bf(f.z) | ((unsigned int)f2bf(f.w) << 16);
  ((uint2*)dst)[i] = r;
}

// ---------------------------------------------------------------------------
// Tiled bf16 WMMA GEMM:  out[N,M] = act(X[N,K] @ W[M,K]^T + bias)
// Block: 256 threads = 8 waves. Tile 128x128, BK=32.
// Wave grid: 4 (M-rows of 32) x 2 (N-cols of 64); each wave 2x4 WMMA tiles.
// Double-buffered LDS fed by GLOBAL_LOAD_ASYNC_TO_LDS_B128 (ASYNCcnt), so the
// next K-tile streams into LDS while the current tile's 8 WMMAs execute.
// ---------------------------------------------------------------------------
#define BM 128
#define BN 128
#define BK 32
#define LDSP 40        // padded LDS row stride (ushorts): 16B aligned, bank skew
#define TILE_USH (2 * BM * LDSP)   // ushorts per (A+B) buffer

#if USE_ASYNC_LDS
#define ASYNC_CP16(gp, lofs) \
  __builtin_amdgcn_global_load_async_to_lds_b128( \
      (AS1 v4i_vs*)(gp), (AS3 v4i_vs*)&SM[lofs], 0, 0)
#endif

__global__ __launch_bounds__(256, 2)
void gemm_bf16_wmma(const unsigned short* __restrict__ X, int ldx,
                    const unsigned short* __restrict__ W, int ldw,
                    const float* __restrict__ bias,
                    unsigned short* __restrict__ outB,
                    int K, int relu, int ldo) {
#if USE_ASYNC_LDS
  __shared__ unsigned short SM[2 * TILE_USH];   // ping-pong buffers
#else
  __shared__ unsigned short SM[TILE_USH];       // single buffer
#endif

  const int tid   = threadIdx.x;
  const int lane  = tid & 31;
  const int wave  = tid >> 5;
  const int waveM = wave & 3;   // 0..3 -> rows of 32
  const int waveN = wave >> 2;  // 0..1 -> cols of 64
  const int rowBase = blockIdx.y * BM;
  const int colBase = blockIdx.x * BN;
  const int half = lane >> 4;
  const int l16  = lane & 15;

  // Staging: thread handles two 16B chunks of X tile and two of W tile.
  const int sr = tid >> 2;   // 0..63
  const int sc = tid & 3;    // 0..3
  const unsigned short* gX0 = X + (size_t)(rowBase + sr) * ldx + sc * 8;
  const unsigned short* gX1 = gX0 + (size_t)64 * ldx;
  const unsigned short* gW0 = W + (size_t)(colBase + sr) * ldw + sc * 8;
  const unsigned short* gW1 = gW0 + (size_t)64 * ldw;
  const int lX0 = sr * LDSP + sc * 8;
  const int lX1 = lX0 + 64 * LDSP;
  const int lW0 = BM * LDSP + sr * LDSP + sc * 8;
  const int lW1 = lW0 + 64 * LDSP;

  v8f acc[2][4];
#pragma unroll
  for (int mi = 0; mi < 2; ++mi)
#pragma unroll
    for (int ni = 0; ni < 4; ++ni)
#pragma unroll
      for (int q = 0; q < 8; ++q) acc[mi][ni][q] = 0.0f;

  const int nIter = K / BK;

#if USE_ASYNC_LDS
  // Prologue: async-stage tile 0 into buffer 0 (no VGPR round trip).
  ASYNC_CP16(gX0, lX0);
  ASYNC_CP16(gX1, lX1);
  ASYNC_CP16(gW0, lW0);
  ASYNC_CP16(gW1, lW1);
#else
  // Prologue: tile 0 into registers.
  uint4 rx0 = *(const uint4*)gX0;
  uint4 rx1 = *(const uint4*)gX1;
  uint4 rw0 = *(const uint4*)gW0;
  uint4 rw1 = *(const uint4*)gW1;
#endif

  for (int it = 0; it < nIter; ++it) {
    const int p = it & 1;
#if USE_ASYNC_LDS
    const unsigned short* Ab = &SM[p * TILE_USH];
    const unsigned short* Bb = Ab + BM * LDSP;
#if __has_builtin(__builtin_amdgcn_s_wait_asynccnt)
    __builtin_amdgcn_s_wait_asynccnt(0);
#else
    asm volatile("s_wait_asynccnt 0x0" ::: "memory");
#endif
    __syncthreads();   // all waves' async tile-`it` writes visible; prior reads done
    if (it + 1 < nIter) {
      const int ko = (it + 1) * BK;
      const int nb = (p ^ 1) * TILE_USH;
      ASYNC_CP16(gX0 + ko, nb + lX0);
      ASYNC_CP16(gX1 + ko, nb + lX1);
      ASYNC_CP16(gW0 + ko, nb + lW0);
      ASYNC_CP16(gW1 + ko, nb + lW1);
    }
#else
    (void)p;
    const unsigned short* Ab = &SM[0];
    const unsigned short* Bb = Ab + BM * LDSP;
    __syncthreads();   // previous iteration's reads done
    *(uint4*)(&SM[lX0]) = rx0;
    *(uint4*)(&SM[lX1]) = rx1;
    *(uint4*)(&SM[lW0]) = rw0;
    *(uint4*)(&SM[lW1]) = rw1;
    __syncthreads();   // tile visible
    if (it + 1 < nIter) {     // prefetch next tile into registers during compute
      const int ko = (it + 1) * BK;
      rx0 = *(const uint4*)(gX0 + ko);
      rx1 = *(const uint4*)(gX1 + ko);
      rw0 = *(const uint4*)(gW0 + ko);
      rw1 = *(const uint4*)(gW1 + ko);
    }
#endif

    // Build fragments per CDNA5 WMMA VGPR layouts.
    // A (16x32 bf16): half=0 -> K pairs {0,2,4,6,16,18,20,22}, half=1 -> +8
    v8i araw[2];
#pragma unroll
    for (int mi = 0; mi < 2; ++mi) {
      const unsigned short* ap = Ab + (waveM * 32 + mi * 16 + l16) * LDSP;
#pragma unroll
      for (int q = 0; q < 8; ++q) {
        int kb = (q < 4) ? (2 * q + 8 * half) : (16 + 2 * (q - 4) + 8 * half);
        araw[mi][q] = *(const int*)(ap + kb);
      }
    }
    // B (32x16 bf16): lane = N column (l16), same K pairs as A (half=1 -> +8)
    v8i braw[4];
#pragma unroll
    for (int ni = 0; ni < 4; ++ni) {
      const unsigned short* bp = Bb + (waveN * 64 + ni * 16 + l16) * LDSP;
#pragma unroll
      for (int q = 0; q < 8; ++q) {
        int kb = (q < 4) ? (2 * q + 8 * half) : (16 + 2 * (q - 4) + 8 * half);
        braw[ni][q] = *(const int*)(bp + kb);
      }
    }

#pragma unroll
    for (int mi = 0; mi < 2; ++mi) {
      v16bf a = __builtin_bit_cast(v16bf, araw[mi]);
#pragma unroll
      for (int ni = 0; ni < 4; ++ni) {
        v16bf b = __builtin_bit_cast(v16bf, braw[ni]);
        acc[mi][ni] = __builtin_amdgcn_wmma_f32_16x16x32_bf16(
            false, a, false, b, (short)0, acc[mi][ni], false, false);
      }
    }
  }

  // Epilogue: C/D layout -> VGPR j: row = j + 8*half, col = lane&15
#pragma unroll
  for (int mi = 0; mi < 2; ++mi) {
#pragma unroll
    for (int ni = 0; ni < 4; ++ni) {
      int col = colBase + waveN * 64 + ni * 16 + l16;
      float bv = bias[col];
      int rbase = rowBase + waveM * 32 + mi * 16 + 8 * half;
#pragma unroll
      for (int j = 0; j < 8; ++j) {
        float v = acc[mi][ni][j] + bv;
        if (relu) v = fmaxf(v, 0.0f);
        outB[(size_t)(rbase + j) * ldo + col] = f2bf(v);
      }
    }
  }
}

// ---------------------------------------------------------------------------
// instance_pred[i][c] = feat[i] . w_fc[c] + b_fc[c]   (wave per row)
// ---------------------------------------------------------------------------
__global__ void ip_kernel(const unsigned short* __restrict__ feat,
                          const float* __restrict__ w_fc,
                          const float* __restrict__ b_fc,
                          float* __restrict__ ip) {
  int gw = (blockIdx.x * blockDim.x + threadIdx.x) >> 5;
  int lane = threadIdx.x & 31;
  const unsigned short* f = feat + (size_t)gw * DMODEL;
  float s0 = 0.f, s1 = 0.f;
  for (int k = lane; k < DMODEL; k += 32) {
    float fv = bf2f(f[k]);
    s0 += fv * w_fc[k];
    s1 += fv * w_fc[DMODEL + k];
  }
#pragma unroll
  for (int off = 16; off > 0; off >>= 1) {
    s0 += __shfl_xor(s0, off, 32);
    s1 += __shfl_xor(s1, off, 32);
  }
  if (lane == 0) {
    ip[(size_t)gw * 2 + 0] = s0 + b_fc[0];
    ip[(size_t)gw * 2 + 1] = s1 + b_fc[1];
  }
}

// ---------------------------------------------------------------------------
// Column max + argmax (first-occurrence ties) over [N,2], two stages
// ---------------------------------------------------------------------------
__global__ void colmax_stage1(const float* __restrict__ mat, int nrows,
                              float* __restrict__ pval, int* __restrict__ pidx) {
  __shared__ float sv0[256], sv1[256];
  __shared__ int si0[256], si1[256];
  int t = threadIdx.x;
  int rpb = nrows / gridDim.x;
  int r0 = blockIdx.x * rpb;
  float bv0 = -3.4e38f, bv1 = -3.4e38f;
  int bi0 = 0x7fffffff, bi1 = 0x7fffffff;
  for (int r = r0 + t; r < r0 + rpb; r += blockDim.x) {
    float v0 = mat[(size_t)r * 2 + 0];
    float v1 = mat[(size_t)r * 2 + 1];
    if (v0 > bv0 || (v0 == bv0 && r < bi0)) { bv0 = v0; bi0 = r; }
    if (v1 > bv1 || (v1 == bv1 && r < bi1)) { bv1 = v1; bi1 = r; }
  }
  sv0[t] = bv0; si0[t] = bi0; sv1[t] = bv1; si1[t] = bi1;
  __syncthreads();
  for (int s = 128; s > 0; s >>= 1) {
    if (t < s) {
      if (sv0[t+s] > sv0[t] || (sv0[t+s] == sv0[t] && si0[t+s] < si0[t])) { sv0[t] = sv0[t+s]; si0[t] = si0[t+s]; }
      if (sv1[t+s] > sv1[t] || (sv1[t+s] == sv1[t] && si1[t+s] < si1[t])) { sv1[t] = sv1[t+s]; si1[t] = si1[t+s]; }
    }
    __syncthreads();
  }
  if (t == 0) {
    pval[blockIdx.x * 2 + 0] = sv0[0]; pidx[blockIdx.x * 2 + 0] = si0[0];
    pval[blockIdx.x * 2 + 1] = sv1[0]; pidx[blockIdx.x * 2 + 1] = si1[0];
  }
}

__global__ void colmax_stage2(const float* __restrict__ pval, const int* __restrict__ pidx,
                              int nb, float* __restrict__ outv, int* __restrict__ outi) {
  __shared__ float sv0[128], sv1[128];
  __shared__ int si0[128], si1[128];
  int t = threadIdx.x;  // blockDim = 128
  if (t < nb) { sv0[t] = pval[t*2+0]; si0[t] = pidx[t*2+0]; sv1[t] = pval[t*2+1]; si1[t] = pidx[t*2+1]; }
  else        { sv0[t] = -3.4e38f; si0[t] = 0x7fffffff; sv1[t] = -3.4e38f; si1[t] = 0x7fffffff; }
  __syncthreads();
  for (int s = 64; s > 0; s >>= 1) {
    if (t < s) {
      if (sv0[t+s] > sv0[t] || (sv0[t+s] == sv0[t] && si0[t+s] < si0[t])) { sv0[t] = sv0[t+s]; si0[t] = si0[t+s]; }
      if (sv1[t+s] > sv1[t] || (sv1[t+s] == sv1[t] && si1[t+s] < si1[t])) { sv1[t] = sv1[t+s]; si1[t] = si1[t+s]; }
    }
    __syncthreads();
  }
  if (t == 0) { outv[0] = sv0[0]; outv[1] = sv1[0]; outi[0] = si0[0]; outi[1] = si1[0]; }
}

// ---------------------------------------------------------------------------
// q_max[c][m] = feat[top_idx[c]] . w_q[m] + b_q[m]   (2048 threads)
// ---------------------------------------------------------------------------
__global__ void qmax_kernel(const unsigned short* __restrict__ feat,
                            const int* __restrict__ top_idx,
                            const float* __restrict__ w_q,
                            const float* __restrict__ b_q,
                            float* __restrict__ q_max) {
  int gid = blockIdx.x * blockDim.x + threadIdx.x;
  if (gid >= 2 * DMODEL) return;
  int c = gid >> 10, m = gid & (DMODEL - 1);
  const unsigned short* f = feat + (size_t)top_idx[c] * DMODEL;
  const float* w = w_q + (size_t)m * DMODEL;
  float s = 0.f;
  for (int k = 0; k < DMODEL; ++k) s += bf2f(f[k]) * w[k];
  q_max[c * DMODEL + m] = s + b_q[m];
}

// ---------------------------------------------------------------------------
// scores[i][c] = (Q[i] . q_max[c]) / 32   (wave per row, Q in bf16)
// ---------------------------------------------------------------------------
__global__ void scores_kernel(const unsigned short* __restrict__ Qb,
                              const float* __restrict__ q_max,
                              float* __restrict__ scores) {
  int gw = (blockIdx.x * blockDim.x + threadIdx.x) >> 5;
  int lane = threadIdx.x & 31;
  const unsigned short* q = Qb + (size_t)gw * DMODEL;
  float s0 = 0.f, s1 = 0.f;
  for (int k = lane; k < DMODEL; k += 32) {
    float qv = bf2f(q[k]);
    s0 += qv * q_max[k];
    s1 += qv * q_max[DMODEL + k];
  }
#pragma unroll
  for (int off = 16; off > 0; off >>= 1) {
    s0 += __shfl_xor(s0, off, 32);
    s1 += __shfl_xor(s1, off, 32);
  }
  if (lane == 0) {
    scores[(size_t)gw * 2 + 0] = s0 * (1.0f / 32.0f);
    scores[(size_t)gw * 2 + 1] = s1 * (1.0f / 32.0f);
  }
}

// ---------------------------------------------------------------------------
// softmax over instances: exp pass (writes unnormalized A) + partial sums
// ---------------------------------------------------------------------------
__global__ void exp_stage1(const float* __restrict__ scores,
                           const float* __restrict__ smax,
                           float* __restrict__ Aout, float* __restrict__ psum) {
  __shared__ float s0a[256], s1a[256];
  int t = threadIdx.x;
  int r = blockIdx.x * blockDim.x + t;
  float e0 = expf(scores[(size_t)r * 2 + 0] - smax[0]);
  float e1 = expf(scores[(size_t)r * 2 + 1] - smax[1]);
  Aout[(size_t)r * 2 + 0] = e0;
  Aout[(size_t)r * 2 + 1] = e1;
  s0a[t] = e0; s1a[t] = e1;
  __syncthreads();
  for (int s = 128; s > 0; s >>= 1) {
    if (t < s) { s0a[t] += s0a[t+s]; s1a[t] += s1a[t+s]; }
    __syncthreads();
  }
  if (t == 0) { psum[blockIdx.x * 2 + 0] = s0a[0]; psum[blockIdx.x * 2 + 1] = s1a[0]; }
}

__global__ void sum_stage2(const float* __restrict__ psum, int nb, float* __restrict__ inv) {
  __shared__ float s0a[128], s1a[128];
  int t = threadIdx.x;  // blockDim = 128
  s0a[t] = (t < nb) ? psum[t*2+0] : 0.f;
  s1a[t] = (t < nb) ? psum[t*2+1] : 0.f;
  __syncthreads();
  for (int s = 64; s > 0; s >>= 1) {
    if (t < s) { s0a[t] += s0a[t+s]; s1a[t] += s1a[t+s]; }
    __syncthreads();
  }
  if (t == 0) { inv[0] = 1.0f / s0a[0]; inv[1] = 1.0f / s1a[0]; }
}

__global__ void norm_kernel(float* __restrict__ A, const float* __restrict__ inv) {
  int g = blockIdx.x * blockDim.x + threadIdx.x;  // 65536
  A[g] *= inv[g & 1];
}

// ---------------------------------------------------------------------------
// B = A^T @ V : chunked partial sums over 256-row chunks, then reduce
// ---------------------------------------------------------------------------
__global__ void wsum_stage1(const float* __restrict__ A,
                            const unsigned short* __restrict__ Vb,
                            float* __restrict__ part) {
  __shared__ float sA[256 * 2];
  int t = threadIdx.x;
  int r0 = blockIdx.x * 256;
  sA[t * 2 + 0] = A[(size_t)(r0 + t) * 2 + 0];
  sA[t * 2 + 1] = A[(size_t)(r0 + t) * 2 + 1];
  __syncthreads();
  int col0 = t * 4;
  float a0acc[4] = {0.f, 0.f, 0.f, 0.f};
  float a1acc[4] = {0.f, 0.f, 0.f, 0.f};
  for (int r = 0; r < 256; ++r) {
    uint2 w = *(const uint2*)(Vb + (size_t)(r0 + r) * DMODEL + col0);
    float f0 = bf2f((unsigned short)(w.x & 0xffff));
    float f1 = bf2f((unsigned short)(w.x >> 16));
    float f2 = bf2f((unsigned short)(w.y & 0xffff));
    float f3 = bf2f((unsigned short)(w.y >> 16));
    float a0 = sA[r * 2 + 0], a1 = sA[r * 2 + 1];
    a0acc[0] += a0 * f0; a0acc[1] += a0 * f1; a0acc[2] += a0 * f2; a0acc[3] += a0 * f3;
    a1acc[0] += a1 * f0; a1acc[1] += a1 * f1; a1acc[2] += a1 * f2; a1acc[3] += a1 * f3;
  }
#pragma unroll
  for (int q = 0; q < 4; ++q) {
    part[(size_t)blockIdx.x * 2048 + 0 * DMODEL + col0 + q] = a0acc[q];
    part[(size_t)blockIdx.x * 2048 + 1 * DMODEL + col0 + q] = a1acc[q];
  }
}

__global__ void wsum_stage2(const float* __restrict__ part,
                            float* __restrict__ outB, float* __restrict__ Bf) {
  int j = blockIdx.x * blockDim.x + threadIdx.x;  // 2048
  float s = 0.f;
  for (int k = 0; k < 128; ++k) s += part[(size_t)k * 2048 + j];
  outB[j] = s;
  Bf[j] = s;
}

// ---------------------------------------------------------------------------
// C[o] = sum_{i,k} w_fcc[o,i,k] * B[i,k] + b_fcc[o]
// ---------------------------------------------------------------------------
__global__ void conv_kernel(const float* __restrict__ w_fcc,
                            const float* __restrict__ Bf,
                            const float* __restrict__ b_fcc,
                            float* __restrict__ Cout) {
  __shared__ float r0[256], r1[256];
  int t = threadIdx.x;
  float p0 = 0.f, p1 = 0.f;
  for (int idx = t; idx < 2048; idx += 256) {
    float b = Bf[idx];
    p0 += w_fcc[idx] * b;
    p1 += w_fcc[2048 + idx] * b;
  }
  r0[t] = p0; r1[t] = p1;
  __syncthreads();
  for (int s = 128; s > 0; s >>= 1) {
    if (t < s) { r0[t] += r0[t+s]; r1[t] += r1[t+s]; }
    __syncthreads();
  }
  if (t == 0) { Cout[0] = r0[0] + b_fcc[0]; Cout[1] = r1[0] + b_fcc[1]; }
}

// ---------------------------------------------------------------------------
// Workspace layout (bytes). V reuses xb, Q reuses hb (dead after GEMM1/2).
// ---------------------------------------------------------------------------
static constexpr size_t SZ_XB   = (size_t)N_INST * D_IN * 2;    // 113,246,208
static constexpr size_t SZ_HB   = (size_t)N_INST * DMODEL * 2;  //  67,108,864
static constexpr size_t OFF_XB   = 0;
static constexpr size_t OFF_HB   = OFF_XB + SZ_XB;
static constexpr size_t OFF_FEAT = OFF_HB + SZ_HB;
static constexpr size_t OFF_WB1  = OFF_FEAT + SZ_HB;
static constexpr size_t OFF_WB2  = OFF_WB1 + (size_t)DMODEL * D_IN * 2;
static constexpr size_t OFF_WQB  = OFF_WB2 + (size_t)DMODEL * DMODEL * 2;
static constexpr size_t OFF_WVB  = OFF_WQB + (size_t)DMODEL * DMODEL * 2;
static constexpr size_t OFF_SCORES = OFF_WVB + (size_t)DMODEL * DMODEL * 2;
static constexpr size_t OFF_PVAL = OFF_SCORES + (size_t)N_INST * 2 * 4;
static constexpr size_t OFF_PIDX = OFF_PVAL + 1024;
static constexpr size_t OFF_RVAL = OFF_PIDX + 1024;
static constexpr size_t OFF_RIDX = OFF_RVAL + 256;
static constexpr size_t OFF_QMAX = OFF_RIDX + 256;
static constexpr size_t OFF_PSUM = OFF_QMAX + 2 * DMODEL * 4;
static constexpr size_t OFF_INV  = OFF_PSUM + 1024;
static constexpr size_t OFF_WPART = OFF_INV + 256;
static constexpr size_t OFF_BF   = OFF_WPART + (size_t)128 * 2 * DMODEL * 4;

extern "C" void kernel_launch(void* const* d_in, const int* in_sizes, int n_in,
                              void* d_out, int out_size, void* d_ws, size_t ws_size,
                              hipStream_t stream) {
  const float* x    = (const float*)d_in[0];
  const float* w1   = (const float*)d_in[1];
  const float* b1   = (const float*)d_in[2];
  const float* w2   = (const float*)d_in[3];
  const float* b2   = (const float*)d_in[4];
  const float* w_fc = (const float*)d_in[5];
  const float* b_fc = (const float*)d_in[6];
  const float* w_q  = (const float*)d_in[7];
  const float* b_q  = (const float*)d_in[8];
  const float* w_v  = (const float*)d_in[9];
  const float* b_v  = (const float*)d_in[10];
  const float* w_fcc= (const float*)d_in[11];
  const float* b_fcc= (const float*)d_in[12];

  char* ws = (char*)d_ws;
  unsigned short* xb   = (unsigned short*)(ws + OFF_XB);
  unsigned short* hb   = (unsigned short*)(ws + OFF_HB);
  unsigned short* featb= (unsigned short*)(ws + OFF_FEAT);
  unsigned short* wb1  = (unsigned short*)(ws + OFF_WB1);
  unsigned short* wb2  = (unsigned short*)(ws + OFF_WB2);
  unsigned short* wqb  = (unsigned short*)(ws + OFF_WQB);
  unsigned short* wvb  = (unsigned short*)(ws + OFF_WVB);
  unsigned short* Vb   = xb;   // reuse: x dead after GEMM1
  unsigned short* Qb   = hb;   // reuse: h dead after GEMM2
  float* scores = (float*)(ws + OFF_SCORES);
  float* pval   = (float*)(ws + OFF_PVAL);
  int*   pidx   = (int*)  (ws + OFF_PIDX);
  float* rval   = (float*)(ws + OFF_RVAL);
  int*   ridx   = (int*)  (ws + OFF_RIDX);
  float* qmax   = (float*)(ws + OFF_QMAX);
  float* psum   = (float*)(ws + OFF_PSUM);
  float* invs   = (float*)(ws + OFF_INV);
  float* wpart  = (float*)(ws + OFF_WPART);
  float* Bf     = (float*)(ws + OFF_BF);

  float* out_ip = (float*)d_out;          // [32768,2]
  float* out_C  = out_ip + 65536;         // [1,2]
  float* out_A  = out_ip + 65538;         // [32768,2]
  float* out_B  = out_ip + 131074;        // [1,2,1024]

  // 1) fp32 -> bf16 conversions
  {
    int n4;
    n4 = N_INST * D_IN / 4;
    cvt_f32_bf16_v4<<<(n4 + 255) / 256, 256, 0, stream>>>(x, xb, n4);
    n4 = DMODEL * D_IN / 4;
    cvt_f32_bf16_v4<<<(n4 + 255) / 256, 256, 0, stream>>>(w1, wb1, n4);
    n4 = DMODEL * DMODEL / 4;
    cvt_f32_bf16_v4<<<(n4 + 255) / 256, 256, 0, stream>>>(w2, wb2, n4);
    cvt_f32_bf16_v4<<<(n4 + 255) / 256, 256, 0, stream>>>(w_q, wqb, n4);
    cvt_f32_bf16_v4<<<(n4 + 255) / 256, 256, 0, stream>>>(w_v, wvb, n4);
  }

  // 2) four big WMMA GEMMs
  dim3 ggrid(DMODEL / BN, N_INST / BM);   // (8, 256)
  gemm_bf16_wmma<<<ggrid, 256, 0, stream>>>(xb, D_IN, wb1, D_IN, b1, hb, D_IN, 1, DMODEL);
  gemm_bf16_wmma<<<ggrid, 256, 0, stream>>>(hb, DMODEL, wb2, DMODEL, b2, featb, DMODEL, 1, DMODEL);
  gemm_bf16_wmma<<<ggrid, 256, 0, stream>>>(featb, DMODEL, wvb, DMODEL, b_v, Vb, DMODEL, 0, DMODEL);
  gemm_bf16_wmma<<<ggrid, 256, 0, stream>>>(featb, DMODEL, wqb, DMODEL, b_q, Qb, DMODEL, 0, DMODEL);

  // 3) instance_pred -> d_out, then argmax over instances
  ip_kernel<<<N_INST / 8, 256, 0, stream>>>(featb, w_fc, b_fc, out_ip);
  colmax_stage1<<<128, 256, 0, stream>>>(out_ip, N_INST, pval, pidx);
  colmax_stage2<<<1, 128, 0, stream>>>(pval, pidx, 128, rval, ridx);

  // 4) q_max for critical instances, attention scores
  qmax_kernel<<<(2 * DMODEL + 255) / 256, 256, 0, stream>>>(featb, ridx, w_q, b_q, qmax);
  scores_kernel<<<N_INST / 8, 256, 0, stream>>>(Qb, qmax, scores);

  // 5) softmax over instances -> A (in d_out)
  colmax_stage1<<<128, 256, 0, stream>>>(scores, N_INST, pval, pidx);
  colmax_stage2<<<1, 128, 0, stream>>>(pval, pidx, 128, rval, ridx);
  exp_stage1<<<128, 256, 0, stream>>>(scores, rval, out_A, psum);
  sum_stage2<<<1, 128, 0, stream>>>(psum, 128, invs);
  norm_kernel<<<65536 / 256, 256, 0, stream>>>(out_A, invs);

  // 6) B = A^T @ V  -> d_out, and C = conv1d(B) -> d_out
  wsum_stage1<<<N_INST / 256, 256, 0, stream>>>(out_A, Vb, wpart);
  wsum_stage2<<<2048 / 256, 256, 0, stream>>>(wpart, out_B, Bf);
  conv_kernel<<<1, 256, 0, stream>>>(w_fcc, Bf, b_fcc, out_C);

  (void)in_sizes; (void)n_in; (void)out_size; (void)ws_size;
}
